// criterion_8830452761380
// MI455X (gfx1250) — matrix operand order repaired and checked by reading, not compiled, not run on hardware
//
#include <hip/hip_runtime.h>
#include <hip/hip_bf16.h>

// ---------------- problem constants (match reference setup_inputs) ----------
constexpr int   L = 4;
constexpr int   C = 12;
constexpr int   S = 96;
constexpr long long P    = (long long)S * S * S;     // 884736 pixels per image
constexpr long long NTOT = (long long)L * C * P;     // 42,467,328 logits
constexpr float SMOOTH = 1e-5f;

// workspace layout (floats)
constexpr int WS_MAXKEY = 0;      // int key of global max (ordered-int encoding)
constexpr int WS_CE     = 1;      // sum of -logp[target]
constexpr int WS_G      = 4;      // 16x16: G[t][c] = sum_{pix: tgt==t} exp(x_c - m); row12 = colsums
constexpr int WS_D1     = WS_G + 256;          // per-l 16x16 prob stats (4 * 256)
constexpr int WS_TOTAL  = WS_D1 + L * 256;

typedef __attribute__((ext_vector_type(16))) _Float16 v16h;
typedef __attribute__((ext_vector_type(8)))  float    v8f;

// monotone float->int key (involution); signed-int compare preserves float order
__device__ __forceinline__ int fkey(int b) { return b >= 0 ? b : (b ^ 0x7FFFFFFF); }

// ---------------- kernel 0: init workspace ---------------------------------
__global__ void k_init(float* ws) {
    int i = blockIdx.x * blockDim.x + threadIdx.x;
    if (i < WS_TOTAL) ws[i] = 0.0f;
    if (i == 0) ((int*)ws)[WS_MAXKEY] = 0x807FFFFF;   // fkey(bits(-inf))
}

// ---------------- kernel 1: global max over all logits ----------------------
__global__ void k_max(const float* __restrict__ x, long long n, int* key) {
    float m = -INFINITY;
    for (long long i = (long long)blockIdx.x * blockDim.x + threadIdx.x;
         i < n; i += (long long)gridDim.x * blockDim.x)
        m = fmaxf(m, x[i]);
    __shared__ float red[256];
    red[threadIdx.x] = m;
    __syncthreads();
    for (int s = 128; s > 0; s >>= 1) {
        if (threadIdx.x < s) red[threadIdx.x] = fmaxf(red[threadIdx.x], red[threadIdx.x + s]);
        __syncthreads();
    }
    if (threadIdx.x == 0) atomicMax(key, fkey(__float_as_int(red[0])));
}

// ---------------- kernel 2: fused main pass ---------------------------------
// grid = (216, L), block = 128 (4 waves). Tiles per image = P/32 = 27648;
// waves per image = 216*4 = 864 -> exactly 32 iterations per wave (uniform).
__global__ void k_main(const float* __restrict__ out, const int* __restrict__ tgt,
                       float* __restrict__ ws) {
    const int l      = blockIdx.y;
    const int tid    = threadIdx.x;
    const int wave   = tid >> 5;
    const int lane   = tid & 31;
    const int laneHi = lane >> 4;     // 0: lanes 0-15, 1: lanes 16-31
    const int lnlo   = lane & 15;

    // decode global max computed by k_max (stream-ordered)
    const int   kk   = __float_as_int(ws[WS_MAXKEY]);
    const float gmax = __int_as_float(fkey(kk));

    // per-wave staging tiles: 32 pixels x 16 cols, f16
    __shared__ _Float16 shP[4][32][16];   // per-pixel softmax probs (+ ones col 12)
    __shared__ _Float16 shE[4][32][16];   // exp(x - gmax)
    __shared__ int      shT[4][32];       // labels
    __shared__ float    red[128];

    v8f acc1 = {};   // D1: OnehotT x Prob  (diag=inter, row12=probsum, col12=count)
    v8f acc2 = {};   // D2: OnehotT x E     (G matrix, row12 = per-class e colsum)
    float ceAcc = 0.0f;

    const int wavesPerL = gridDim.x * (blockDim.x >> 5);
    const int gwave     = blockIdx.x * (blockDim.x >> 5) + wave;
    const int numTiles  = (int)(P / 32);
    const float* baseO  = out + (size_t)l * C * P;
    const int*   baseT  = tgt + (size_t)l * P;

    for (int tile = gwave; tile < numTiles; tile += wavesPerL) {
        const long long p = (long long)tile * 32 + lane;

        // prefetch next tile's class-0 stream (global_prefetch_b8)
        {
            long long pn = (long long)(tile + wavesPerL) * 32 + lane;
            if (pn < P) __builtin_prefetch(baseO + pn, 0, 0);
        }

        float x[C];
        #pragma unroll
        for (int c = 0; c < C; ++c) x[c] = baseO[(long long)c * P + p];

        // per-pixel softmax (stable)
        float pm = x[0];
        #pragma unroll
        for (int c = 1; c < C; ++c) pm = fmaxf(pm, x[c]);
        float e[C], se = 0.0f;
        #pragma unroll
        for (int c = 0; c < C; ++c) { e[c] = __expf(x[c] - pm); se += e[c]; }
        const float inv = __frcp_rn(se);
        const float lse = pm + __logf(se);
        const int   t   = baseT[p];
        ceAcc -= (x[t] - lse);

        // stage f16 tiles (cols 12..15 padding; prob col 12 = 1 for counts)
        #pragma unroll
        for (int c = 0; c < C; ++c) {
            shP[wave][lane][c] = (_Float16)(e[c] * inv);
            shE[wave][lane][c] = (_Float16)__expf(x[c] - gmax);
        }
        shP[wave][lane][12] = (_Float16)1.0f;
        #pragma unroll
        for (int c = 13; c < 16; ++c) shP[wave][lane][c] = (_Float16)0.0f;
        #pragma unroll
        for (int c = 12; c < 16; ++c) shE[wave][lane][c] = (_Float16)0.0f;
        shT[wave][lane] = t;
        __syncthreads();

        // ---- pack WMMA fragments (wave32; contiguous k, same order in A and B) ----
        // A 16x32 f16: lane m (+16): row m; elem j -> K = j + 16*laneHi
        // B 32x16 f16: lane n (+16): col n; elem j -> K = j + 16*laneHi
        v16h a, bp, be;
        const int m = lnlo;
        #pragma unroll
        for (int j = 0; j < 16; ++j) {
            const int ka = j + 16 * laneHi;
            const int tk = shT[wave][ka];
            a[j] = (_Float16)(((m == tk) || (m == 12)) ? 1.0f : 0.0f);
            const int kb = j + 16 * laneHi;
            bp[j] = shP[wave][kb][lnlo];
            be[j] = shE[wave][kb][lnlo];
        }
        acc1 = __builtin_amdgcn_wmma_f32_16x16x32_f16(false, a, false, bp,
                                                      (short)0, acc1, false, false);
        acc2 = __builtin_amdgcn_wmma_f32_16x16x32_f16(false, a, false, be,
                                                      (short)0, acc2, false, false);
        __syncthreads();
    }

    // merge accumulators: D/C layout: VGPR v -> row v + 8*laneHi, col = lane%16
    #pragma unroll
    for (int v = 0; v < 8; ++v) {
        const int row = v + 8 * laneHi;
        const int col = lnlo;
        unsafeAtomicAdd(&ws[WS_D1 + l * 256 + row * 16 + col], acc1[v]);
        unsafeAtomicAdd(&ws[WS_G  +           row * 16 + col], acc2[v]);
    }

    red[tid] = ceAcc;
    __syncthreads();
    for (int s = 64; s > 0; s >>= 1) {
        if (tid < s) red[tid] += red[tid + s];
        __syncthreads();
    }
    if (tid == 0) unsafeAtomicAdd(&ws[WS_CE], red[0]);
}

// ---------------- kernel 3: finalize to scalar -------------------------------
__global__ void k_fin(const float* __restrict__ M, const float* __restrict__ ws,
                      float* __restrict__ o) {
    if (blockIdx.x != 0 || threadIdx.x != 0) return;

    const float* G = ws + WS_G;
    float gs = 0.0f;                              // global sum of exp(x - m)
    for (int c = 0; c < C; ++c) gs += G[12 * 16 + c];
    float pen = 0.0f;
    for (int t = 0; t < C; ++t)
        for (int c = 0; c < C; ++c) pen += M[t * C + c] * G[t * 16 + c];
    pen /= gs;

    const float ce = ws[WS_CE] / (float)((long long)L * P);

    float dice = 0.0f;
    for (int l = 0; l < L; ++l) {
        const float* D = ws + WS_D1 + l * 256;
        for (int c = 0; c < C; ++c) {
            const float inter   = D[c * 16 + c];
            const float probsum = D[12 * 16 + c];
            const float cnt     = D[c * 16 + 12];
            dice += 1.0f - (2.0f * inter + SMOOTH) / (probsum + cnt + SMOOTH);
        }
    }
    dice /= (float)(L * C);

    o[0] = dice + ce + pen / (float)((long long)L * P);
}

// ---------------- launcher ---------------------------------------------------
extern "C" void kernel_launch(void* const* d_in, const int* in_sizes, int n_in,
                              void* d_out, int out_size, void* d_ws, size_t ws_size,
                              hipStream_t stream) {
    const float* outputs = (const float*)d_in[0];
    const int*   targets = (const int*)d_in[1];
    const float* penalty = (const float*)d_in[2];
    float* ws = (float*)d_ws;
    float* o  = (float*)d_out;

    k_init<<<(WS_TOTAL + 255) / 256, 256, 0, stream>>>(ws);
    k_max<<<1024, 256, 0, stream>>>(outputs, NTOT, (int*)ws);
    dim3 grid(216, L);
    k_main<<<grid, 128, 0, stream>>>(outputs, targets, ws);
    k_fin<<<1, 32, 0, stream>>>(penalty, ws, o);
}
